// SlimmableMoE_8366596292693
// MI455X (gfx1250) — matrix operand order repaired and checked by reading, not compiled, not run on hardware
//
#include <hip/hip_runtime.h>
#include <hip/hip_bf16.h>
#include <math.h>

// ---------------- problem constants ----------------
enum : int {
  E_ = 8, B_ = 4, S_ = 1024, D_ = 1024, F_ = 4096,
  W_ = 768, H_ = 6, HD_ = 128, M_ = B_ * S_   // 4096 tokens
};

typedef __bf16 bf16;
typedef __attribute__((ext_vector_type(16))) __bf16 v16bf;
typedef __attribute__((ext_vector_type(8)))  __bf16 bf16x8;
typedef __attribute__((ext_vector_type(8)))  float  v8f;

// generic-pointer low 32 bits == LDS byte address (LDS aperture truncation)
__device__ __forceinline__ unsigned lds_addr32(const void* p) {
  return (unsigned)(unsigned long long)p;
}

// async copy of 32 bytes: global -> LDS (two b128 ops; IOFFSET advances both sides)
__device__ __forceinline__ void async_copy_b256(unsigned lds, const void* gptr) {
  asm volatile(
      "global_load_async_to_lds_b128 %0, %1, off\n\t"
      "global_load_async_to_lds_b128 %0, %1, off offset:16"
      :: "v"(lds), "v"(gptr) : "memory");
}

__device__ __forceinline__ void wait_async_all() {
  asm volatile("s_wait_asynccnt 0x0" ::: "memory");
}

// ---------------- WMMA fragment loaders (CDNA5 bf16 16x16x32) ----------------
// A matrix 16x32 (MxK): lanes 0-15 rows M=0..15 hold K {0..7, 16..23};
// lanes 16-31 rows M=0..15 hold K {8..15, 24..31}. (ISA 7.12.2)
__device__ __forceinline__ v16bf load_frag_a(const bf16* __restrict__ p, int ld) {
  const int lane = threadIdx.x & 31;
  const int half = lane >> 4;
  const int r    = lane & 15;
  const bf16* row = p + r * ld + half * 8;
  bf16x8 lo = *(const bf16x8*)(row);
  bf16x8 hi = *(const bf16x8*)(row + 16);
  v16bf f;
#pragma unroll
  for (int i = 0; i < 8; ++i) { f[i] = lo[i]; f[i + 8] = hi[i]; }
  return f;
}

// B matrix 32x16 (KxN), source stored row-major [N][K]:
// lanes 0-15: column N=lane, K {0..7, 16..23} ; lanes 16-31: column N=lane-16, K {8..15, 24..31}.
__device__ __forceinline__ v16bf load_frag_b(const bf16* __restrict__ p, int ld) {
  const int lane = threadIdx.x & 31;
  const int half = lane >> 4;
  const int n    = lane & 15;
  const bf16* row = p + n * ld + half * 8;
  bf16x8 lo = *(const bf16x8*)(row);
  bf16x8 hi = *(const bf16x8*)(row + 16);
  v16bf f;
#pragma unroll
  for (int i = 0; i < 8; ++i) { f[i] = lo[i]; f[i + 8] = hi[i]; }
  return f;
}

// ---------------- generic slice fp32 -> bf16 convert ----------------
__global__ void cvt_slice_bf16(const float* __restrict__ src, bf16* __restrict__ dst,
                               int R, int C, int ld) {
  size_t n = (size_t)R * C;
  for (size_t i = (size_t)blockIdx.x * blockDim.x + threadIdx.x; i < n;
       i += (size_t)gridDim.x * blockDim.x) {
    size_t r = i / (size_t)C, c = i - r * (size_t)C;
    dst[i] = (bf16)src[r * (size_t)ld + c];
  }
}

// ---------------- zero init of d_out ----------------
__global__ void zero_f32(float* __restrict__ p, size_t n) {
  for (size_t i = (size_t)blockIdx.x * blockDim.x + threadIdx.x; i < n;
       i += (size_t)gridDim.x * blockDim.x) p[i] = 0.0f;
}

// ---------------- router: softmax + top-2 renormalized gate ----------------
__global__ void router_kernel(const float* __restrict__ x, const float* __restrict__ wr,
                              float* __restrict__ gate) {
  int t = blockIdx.x * blockDim.x + threadIdx.x;
  if (t >= M_) return;
  const float* xr = x + (size_t)t * D_;
  float lg[E_];
#pragma unroll
  for (int e = 0; e < E_; ++e) lg[e] = 0.0f;
  for (int d = 0; d < D_; ++d) {
    float xv = xr[d];
#pragma unroll
    for (int e = 0; e < E_; ++e) lg[e] += xv * wr[d * E_ + e];
  }
  float mx = lg[0];
#pragma unroll
  for (int e = 1; e < E_; ++e) mx = fmaxf(mx, lg[e]);
  float s = 0.0f;
#pragma unroll
  for (int e = 0; e < E_; ++e) { lg[e] = __expf(lg[e] - mx); s += lg[e]; }
#pragma unroll
  for (int e = 0; e < E_; ++e) lg[e] /= s;
  int i1 = 0;
#pragma unroll
  for (int e = 1; e < E_; ++e) if (lg[e] > lg[i1]) i1 = e;
  int i2 = (i1 == 0) ? 1 : 0;
#pragma unroll
  for (int e = 0; e < E_; ++e) if (e != i1 && lg[e] > lg[i2]) i2 = e;
  float tot = lg[i1] + lg[i2];
  float g[E_];
#pragma unroll
  for (int e = 0; e < E_; ++e) g[e] = 0.0f;
  g[i1] = lg[i1] / tot;
  g[i2] = lg[i2] / tot;
#pragma unroll
  for (int e = 0; e < E_; ++e) gate[(size_t)t * E_ + e] = g[e];
}

// ---------------- tiled WMMA GEMM: out[M,N] = A[M,K] @ Wt[N,K]^T + bias ----------------
// Double-buffered LDS; tiles staged with global_load_async_to_lds_b128 (ASYNCcnt).
// flags bit0: output bf16 (else f32), bit1: exact GELU epilogue.
#define GBM 128
#define GBN 128
#define GBK 32
#define GLD 40   // 32 + 8 pad (bf16 elems); row stride 80B keeps 16B alignment
__global__ __launch_bounds__(256)
void gemm_bf16_wmma(const bf16* __restrict__ A, const bf16* __restrict__ Bw,
                    const float* __restrict__ bias, void* __restrict__ out,
                    int M, int N, int K, int flags) {
  __shared__ __align__(16) bf16 As[2][GBM * GLD];
  __shared__ __align__(16) bf16 Bs[2][GBN * GLD];
  const int m0 = blockIdx.x * GBM;
  const int n0 = blockIdx.y * GBN;
  const int tid  = threadIdx.x;
  const int wave = tid >> 5, lane = tid & 31;
  const int wm = wave & 3;   // 0..3  -> 32-row slab
  const int wn = wave >> 2;  // 0..1  -> 64-col slab

  v8f c[2][4];
#pragma unroll
  for (int i = 0; i < 2; ++i)
#pragma unroll
    for (int j = 0; j < 4; ++j) c[i][j] = (v8f){};

  const int ldrow = tid >> 1;          // 0..127
  const int ldseg = (tid & 1) * 16;    // 0 or 16
  const bf16* aptr = A  + (size_t)(m0 + ldrow) * K + ldseg;
  const bf16* bptr = Bw + (size_t)(n0 + ldrow) * K + ldseg;
  const unsigned la0 = lds_addr32(&As[0][ldrow * GLD + ldseg]);
  const unsigned la1 = lds_addr32(&As[1][ldrow * GLD + ldseg]);
  const unsigned lb0 = lds_addr32(&Bs[0][ldrow * GLD + ldseg]);
  const unsigned lb1 = lds_addr32(&Bs[1][ldrow * GLD + ldseg]);

  // prologue: stage k-tile 0 into buffer 0
  async_copy_b256(la0, aptr);
  async_copy_b256(lb0, bptr);
  wait_async_all();
  __syncthreads();

  int buf = 0;
  for (int k0 = 0; k0 < K; k0 += GBK) {
    const int nbuf = buf ^ 1;
    if (k0 + GBK < K) {   // stage next tile while WMMAs run on current one
      async_copy_b256(nbuf ? la1 : la0, aptr + k0 + GBK);
      async_copy_b256(nbuf ? lb1 : lb0, bptr + k0 + GBK);
    }

    const bf16* Ab = As[buf];
    const bf16* Bb = Bs[buf];
    v16bf af[2], bfg[4];
    af[0] = load_frag_a(&Ab[(wm * 32     ) * GLD], GLD);
    af[1] = load_frag_a(&Ab[(wm * 32 + 16) * GLD], GLD);
#pragma unroll
    for (int j = 0; j < 4; ++j)
      bfg[j] = load_frag_b(&Bb[(wn * 64 + j * 16) * GLD], GLD);
#pragma unroll
    for (int i = 0; i < 2; ++i)
#pragma unroll
      for (int j = 0; j < 4; ++j)
        c[i][j] = __builtin_amdgcn_wmma_f32_16x16x32_bf16(
            false, af[i], false, bfg[j], (short)0, c[i][j], false, false);

    wait_async_all();      // my next-tile copies have landed
    __syncthreads();       // everyone done reading current buffer
    buf = nbuf;
  }

  // epilogue: C layout -> row = vgpr + 8*(lane>=16), col = lane&15
  const int col   = lane & 15;
  const int rbase = (lane >> 4) * 8;
  const bool out_bf = (flags & 1) != 0;
  const bool gelu   = (flags & 2) != 0;
#pragma unroll
  for (int i = 0; i < 2; ++i) {
#pragma unroll
    for (int j = 0; j < 4; ++j) {
      const int gn = n0 + wn * 64 + j * 16 + col;
      const float bv = bias[gn];
#pragma unroll
      for (int r = 0; r < 8; ++r) {
        const int gm = m0 + wm * 32 + i * 16 + rbase + r;
        float v = c[i][j][r] + bv;
        if (gelu) v = 0.5f * v * (1.0f + erff(v * 0.70710678118654752f));
        size_t idx = (size_t)gm * N + gn;
        if (out_bf) ((bf16*)out)[idx] = (bf16)v;
        else        ((float*)out)[idx] = v;
      }
    }
  }
}

// ---------------- flash attention (per expert): 64q x 64k tiles, HD=128 ----------------
#define AQ 64
#define AKT 64
__global__ __launch_bounds__(128)
void attn_wmma(const bf16* __restrict__ Q, const bf16* __restrict__ Kg,
               const bf16* __restrict__ Vg, bf16* __restrict__ AO) {
  __shared__ __align__(16) bf16  Ks[AKT][HD_ + 8];     // keys x d      (17.4 KB)
  __shared__ __align__(16) bf16  Vt[HD_][AKT + 8];     // d x keys      (18.4 KB)
  __shared__ float              Sc[AQ][AKT];           // scores f32    (16 KB)
  __shared__ __align__(16) bf16  Pb[AQ][AKT + 8];      // probs bf16    ( 9.2 KB)
  __shared__ float              rowscale[AQ];

  const int qt = blockIdx.x;            // query tile
  const int bh = blockIdx.y;            // b*H + h
  const int b = bh / H_, h = bh % H_;
  const int tid  = threadIdx.x;
  const int wave = tid >> 5, lane = tid & 31;
  const size_t baserow = (size_t)b * S_;
  const int q0 = qt * AQ;

  // Q fragments for this wave's 16 query rows (kept in registers)
  v16bf qa[4];
  {
    const bf16* qbase = Q + (baserow + q0 + wave * 16) * W_ + h * HD_;
#pragma unroll
    for (int ks = 0; ks < 4; ++ks) qa[ks] = load_frag_a(qbase + ks * 32, W_);
  }

  v8f oacc[8];
#pragma unroll
  for (int t = 0; t < 8; ++t) oacc[t] = (v8f){};
  float m_i = -3.0e38f, l_i = 0.0f;                 // rows owned by tid<64
  const float scale = 0.08838834764831845f;          // 1/sqrt(128)

  const int krow = tid >> 1;
  const int seg  = (tid & 1) * 64;
  const unsigned lds_k = lds_addr32(&Ks[krow][seg]);

  for (int j0 = 0; j0 < S_; j0 += AKT) {
    // K tile: async global->LDS (128 contiguous bytes per thread, 8 b128 ops)
    {
      const bf16* kb = Kg + (baserow + j0 + krow) * W_ + h * HD_ + seg;
#pragma unroll
      for (int i = 0; i < 4; ++i)
        async_copy_b256(lds_k + i * 32, kb + i * 16);
      // V tile: transpose through VGPRs into Vt
      const bf16* vb = Vg + (baserow + j0 + krow) * W_ + h * HD_ + seg;
#pragma unroll
      for (int i = 0; i < 8; ++i) {
        bf16x8 vv = *(const bf16x8*)(vb + i * 8);
#pragma unroll
        for (int u = 0; u < 8; ++u) Vt[seg + i * 8 + u][krow] = vv[u];
      }
    }
    wait_async_all();
    __syncthreads();

    // scores: wave computes rows [wave*16, +16) x 64 keys via WMMA
    {
      const int col = lane & 15, rbase = (lane >> 4) * 8;
#pragma unroll
      for (int nt = 0; nt < 4; ++nt) {
        v8f acc = (v8f){};
#pragma unroll
        for (int ks = 0; ks < 4; ++ks) {
          v16bf bk = load_frag_b(&Ks[nt * 16][ks * 32], HD_ + 8);
          acc = __builtin_amdgcn_wmma_f32_16x16x32_bf16(
              false, qa[ks], false, bk, (short)0, acc, false, false);
        }
#pragma unroll
        for (int r = 0; r < 8; ++r)
          Sc[wave * 16 + rbase + r][nt * 16 + col] = acc[r] * scale;
      }
    }
    __syncthreads();

    // online softmax: one thread per query row
    if (tid < AQ) {
      float mx = m_i;
#pragma unroll 8
      for (int cidx = 0; cidx < AKT; ++cidx) mx = fmaxf(mx, Sc[tid][cidx]);
      float alpha = __expf(m_i - mx);
      float s = 0.0f;
#pragma unroll 8
      for (int cidx = 0; cidx < AKT; ++cidx) {
        float p = __expf(Sc[tid][cidx] - mx);
        Pb[tid][cidx] = (bf16)p;
        s += p;
      }
      l_i = l_i * alpha + s;
      m_i = mx;
      rowscale[tid] = alpha;
    }
    __syncthreads();

    // rescale O and accumulate P @ V via WMMA
    {
      const int rbase = (lane >> 4) * 8;
      float al[8];
#pragma unroll
      for (int r = 0; r < 8; ++r) al[r] = rowscale[wave * 16 + rbase + r];
#pragma unroll
      for (int nt = 0; nt < 8; ++nt)
#pragma unroll
        for (int r = 0; r < 8; ++r) oacc[nt][r] *= al[r];
      v16bf pa[2];
      pa[0] = load_frag_a(&Pb[wave * 16][0],  AKT + 8);
      pa[1] = load_frag_a(&Pb[wave * 16][32], AKT + 8);
#pragma unroll
      for (int nt = 0; nt < 8; ++nt) {
#pragma unroll
        for (int ks = 0; ks < 2; ++ks) {
          v16bf bv = load_frag_b(&Vt[nt * 16][ks * 32], AKT + 8);
          oacc[nt] = __builtin_amdgcn_wmma_f32_16x16x32_bf16(
              false, pa[ks], false, bv, (short)0, oacc[nt], false, false);
        }
      }
    }
    __syncthreads();
  }

  // final 1/l normalization and store
  if (tid < AQ) rowscale[tid] = 1.0f / l_i;
  __syncthreads();
  {
    const int col = lane & 15, rbase = (lane >> 4) * 8;
#pragma unroll
    for (int r = 0; r < 8; ++r) {
      const float inv = rowscale[wave * 16 + rbase + r];
      const size_t grow = (baserow + q0 + wave * 16 + rbase + r) * W_ + h * HD_;
#pragma unroll
      for (int nt = 0; nt < 8; ++nt)
        AO[grow + nt * 16 + col] = (bf16)(oacc[nt][r] * inv);
    }
  }
}

// ---------------- fused add + RMSNorm (+optional gated accumulate into d_out) ----------
// mode 0: y=rmsnorm(a+b) -> outF (f32, W cols) and outB (bf16)
// mode 1: d_out[row*D + c] += gate[row*E+e] * rmsnorm(a+b)
__global__ __launch_bounds__(256)
void add_rmsnorm(const float* __restrict__ A, int lda, const float* __restrict__ Bv,
                 float* __restrict__ outF, bf16* __restrict__ outB,
                 const float* __restrict__ gate, int e, float* __restrict__ acc,
                 int mode) {
  __shared__ float red[256];
  __shared__ float s_inv;
  const int row = blockIdx.x;
  const float* a = A  + (size_t)row * lda;
  const float* bb = Bv + (size_t)row * W_;
  float ss = 0.0f;
  for (int c = threadIdx.x; c < W_; c += 256) {
    float t = a[c] + bb[c];
    ss += t * t;
  }
  red[threadIdx.x] = ss;
  __syncthreads();
  for (int off = 128; off > 0; off >>= 1) {
    if (threadIdx.x < off) red[threadIdx.x] += red[threadIdx.x + off];
    __syncthreads();
  }
  if (threadIdx.x == 0) s_inv = rsqrtf(red[0] / (float)W_ + 1e-6f);
  __syncthreads();
  const float inv = s_inv;
  if (mode == 0) {
    for (int c = threadIdx.x; c < W_; c += 256) {
      float y = (a[c] + bb[c]) * inv;
      outF[(size_t)row * W_ + c] = y;
      outB[(size_t)row * W_ + c] = (bf16)y;
    }
  } else {
    const float g = gate[(size_t)row * E_ + e];
    for (int c = threadIdx.x; c < W_; c += 256) {
      float y = (a[c] + bb[c]) * inv;
      acc[(size_t)row * D_ + c] += g * y;
    }
  }
}

// ---------------- host orchestration ----------------
extern "C" void kernel_launch(void* const* d_in, const int* in_sizes, int n_in,
                              void* d_out, int out_size, void* d_ws, size_t ws_size,
                              hipStream_t stream) {
  const float* x  = (const float*)d_in[0];
  const float* wr = (const float*)d_in[1];
  const float* qw = (const float*)d_in[2];
  const float* qb = (const float*)d_in[3];
  const float* kw = (const float*)d_in[4];
  const float* kb = (const float*)d_in[5];
  const float* vw = (const float*)d_in[6];
  const float* vb = (const float*)d_in[7];
  const float* ow = (const float*)d_in[8];
  const float* ob = (const float*)d_in[9];
  const float* w1 = (const float*)d_in[10];
  const float* b1 = (const float*)d_in[11];
  const float* w2 = (const float*)d_in[12];
  const float* b2 = (const float*)d_in[13];
  float* out = (float*)d_out;

  // workspace carve-up
  char* ws = (char*)d_ws;
  size_t off = 0;
  auto carve = [&](size_t bytes) -> char* {
    char* p = ws + off;
    off = (off + bytes + 255) & ~(size_t)255;
    return p;
  };
  float* gate  = (float*)carve((size_t)M_ * E_ * 4);
  bf16* xs_bf  = (bf16*)carve((size_t)M_ * W_ * 2);
  bf16* wq_bf  = (bf16*)carve((size_t)W_ * W_ * 2);
  bf16* wk_bf  = (bf16*)carve((size_t)W_ * W_ * 2);
  bf16* wv_bf  = (bf16*)carve((size_t)W_ * W_ * 2);
  bf16* wo_bf  = (bf16*)carve((size_t)W_ * W_ * 2);
  bf16* w1_bf  = (bf16*)carve((size_t)F_ * W_ * 2);
  bf16* w2_bf  = (bf16*)carve((size_t)W_ * F_ * 2);
  bf16* q_bf   = (bf16*)carve((size_t)M_ * W_ * 2);
  bf16* k_bf   = (bf16*)carve((size_t)M_ * W_ * 2);
  bf16* v_bf   = (bf16*)carve((size_t)M_ * W_ * 2);
  bf16* ao_bf  = (bf16*)carve((size_t)M_ * W_ * 2);
  float* t_f32 = (float*)carve((size_t)M_ * W_ * 4);   // o-proj out, reused for FFN2 out
  float* x1_f  = (float*)carve((size_t)M_ * W_ * 4);
  bf16* x1_bf  = (bf16*)carve((size_t)M_ * W_ * 2);
  bf16* h_bf   = (bf16*)carve((size_t)M_ * F_ * 2);

  // zero output (incl. padding cols 768..1023)
  zero_f32<<<2048, 256, 0, stream>>>(out, (size_t)M_ * D_);

  // router gate
  router_kernel<<<(M_ + 255) / 256, 256, 0, stream>>>(x, wr, gate);

  // x_s -> bf16 (slice first 768 cols of D=1024)
  cvt_slice_bf16<<<2048, 256, 0, stream>>>(x, xs_bf, M_, W_, D_);

  const dim3 gemm_wn(M_ / GBM, W_ / GBN);   // 32 x 6
  const dim3 gemm_fn(M_ / GBM, F_ / GBN);   // 32 x 32
  const dim3 attn_grid(S_ / AQ, B_ * H_);   // 16 x 24

  for (int e = 0; e < E_; ++e) {
    const size_t dd = (size_t)D_ * D_;
    const size_t fd = (size_t)F_ * D_;
    // weight slices -> bf16 [N,K] row-major
    cvt_slice_bf16<<<2048, 256, 0, stream>>>(qw + e * dd, wq_bf, W_, W_, D_);
    cvt_slice_bf16<<<2048, 256, 0, stream>>>(kw + e * dd, wk_bf, W_, W_, D_);
    cvt_slice_bf16<<<2048, 256, 0, stream>>>(vw + e * dd, wv_bf, W_, W_, D_);
    cvt_slice_bf16<<<2048, 256, 0, stream>>>(ow + e * dd, wo_bf, W_, W_, D_);
    cvt_slice_bf16<<<2048, 256, 0, stream>>>(w1 + e * fd, w1_bf, F_, W_, D_);
    cvt_slice_bf16<<<2048, 256, 0, stream>>>(w2 + e * fd, w2_bf, W_, F_, F_);

    // Q,K,V projections (bf16 out)
    gemm_bf16_wmma<<<gemm_wn, 256, 0, stream>>>(xs_bf, wq_bf, qb + e * D_, q_bf, M_, W_, W_, 1);
    gemm_bf16_wmma<<<gemm_wn, 256, 0, stream>>>(xs_bf, wk_bf, kb + e * D_, k_bf, M_, W_, W_, 1);
    gemm_bf16_wmma<<<gemm_wn, 256, 0, stream>>>(xs_bf, wv_bf, vb + e * D_, v_bf, M_, W_, W_, 1);

    // attention
    attn_wmma<<<attn_grid, 128, 0, stream>>>(q_bf, k_bf, v_bf, ao_bf);

    // output projection (f32 out)
    gemm_bf16_wmma<<<gemm_wn, 256, 0, stream>>>(ao_bf, wo_bf, ob + e * D_, t_f32, M_, W_, W_, 0);

    // x1 = rmsnorm(x_s + ao)
    add_rmsnorm<<<M_, 256, 0, stream>>>(x, D_, t_f32, x1_f, x1_bf, nullptr, 0, nullptr, 0);

    // FFN: h = gelu(x1 @ w1^T + b1)   (bf16 out), f = h @ w2^T + b2 (f32 out)
    gemm_bf16_wmma<<<gemm_fn, 256, 0, stream>>>(x1_bf, w1_bf, b1 + e * F_, h_bf, M_, F_, W_, 1 | 2);
    gemm_bf16_wmma<<<gemm_wn, 256, 0, stream>>>(h_bf, w2_bf, b2 + e * D_, t_f32, M_, W_, F_, 0);

    // out += gate[e] * rmsnorm(x1 + f)
    add_rmsnorm<<<M_, 256, 0, stream>>>(x1_f, W_, t_f32, nullptr, nullptr, gate, e, out, 1);
  }
}
